// SpatialAudioRenderer_15290083574015
// MI455X (gfx1250) — compile-verified, matrix operand order checked by reading, not run on hardware
//
#include <hip/hip_runtime.h>
#include <hip/hip_bf16.h>
#include <math.h>

typedef __attribute__((ext_vector_type(16))) _Float16 v16h;
typedef __attribute__((ext_vector_type(2)))  _Float16 v2h;
typedef __attribute__((ext_vector_type(8)))  float    v8f;

constexpr int TLEN = 131072;   // time samples
constexpr int NB   = 8;        // batch
// conv(64->128,K=7) as 7 taps x (K=64 = 2 WMMA k-steps) = 14 fragments
constexpr int HSTR = 72;       // padded row stride of hT panel (f16): 36 dwords,
                               // n*36 mod 64 distinct for n=0..15 -> conflict-free

// ---------------------------------------------------------------------------
// conv(2->64, K=7) over full signal, accumulate per-channel sum / sumsq
// ---------------------------------------------------------------------------
__global__ __launch_bounds__(256) void
k_conv_stats(const float* __restrict__ in,
             const float* __restrict__ Wk,
             const float* __restrict__ bk,
             float* __restrict__ sumOut,
             float* __restrict__ sqOut)
{
  __shared__ float xs[2][2056];
  __shared__ float ws[64 * 14];
  __shared__ float bs[64];
  __shared__ float redS[256], redQ[256];
  const int tid  = threadIdx.x;
  const int b    = blockIdx.x >> 6;          // 64 chunks of 2048 per batch
  const int base = (blockIdx.x & 63) * 2048;

  for (int v = tid; v < 64 * 14; v += 256) ws[v] = Wk[v];
  if (tid < 64) bs[tid] = bk[tid];
  for (int v = tid; v < 2 * 2054; v += 256) {
    int i2 = v / 2054, j = v - i2 * 2054;
    int gt = base - 3 + j;
    xs[i2][j] = (gt >= 0 && gt < TLEN) ? in[((size_t)b * 2 + i2) * TLEN + gt] : 0.0f;
  }
  __syncthreads();

  const int c = tid & 63;
  const int s = tid >> 6;
  float aS = 0.0f, aQ = 0.0f;
  for (int tw = s; tw < 2048; tw += 4) {
    float y = bs[c];
#pragma unroll
    for (int i2 = 0; i2 < 2; ++i2)
#pragma unroll
      for (int k = 0; k < 7; ++k)
        y += ws[c * 14 + i2 * 7 + k] * xs[i2][tw + k];
    aS += y; aQ += y * y;
  }
  redS[tid] = aS; redQ[tid] = aQ;
  __syncthreads();
  if (tid < 64) {
    float tS = redS[tid] + redS[tid + 64] + redS[tid + 128] + redS[tid + 192];
    float tQ = redQ[tid] + redQ[tid + 64] + redQ[tid + 128] + redQ[tid + 192];
    atomicAdd(&sumOut[tid], tS);
    atomicAdd(&sqOut[tid], tQ);
  }
}

// ---------------------------------------------------------------------------
// finalize BN: scale = g * rsqrt(var+eps), shift = beta - mean*scale
// ---------------------------------------------------------------------------
__global__ __launch_bounds__(128) void
k_finalize(const float* sum, const float* sq,
           const float* gma, const float* bta,
           float* scale, float* shift, int C, float invN)
{
  int c = threadIdx.x;
  if (c < C) {
    float mean = sum[c] * invN;
    float var  = sq[c] * invN - mean * mean;
    float s    = gma[c] * rsqrtf(var + 1e-5f);
    scale[c] = s;
    shift[c] = bta[c] - mean * s;
  }
}

// ---------------------------------------------------------------------------
// repack W (128 x 64 x 7 f32) into per-tap WMMA A-fragment f16 layout:
// [mtile 0..7][frag 0..13][lane 0..31][e 0..15], frag = dk*2 + kstep
// A element (lane,e) = W[m][ci][dk], ci = kstep*32 + kb + e (+8 if e>=8)
// ---------------------------------------------------------------------------
__global__ __launch_bounds__(256) void
k_pack_wfrag(const float* __restrict__ W, _Float16* __restrict__ outF)
{
  int idx = blockIdx.x * 256 + threadIdx.x;
  if (idx >= 8 * 14 * 512) return;
  int e    = idx & 15;
  int lane = (idx >> 4) & 31;
  int q    = idx >> 9;             // mt*14 + frag
  int frag = q % 14, mt = q / 14;
  int dk   = frag >> 1, kst = frag & 1;
  int m    = mt * 16 + (lane & 15);
  int kb   = (lane >> 4) * 8;
  int ci   = kst * 32 + kb + e + ((e >= 8) ? 8 : 0);
  outF[idx] = (_Float16)W[(m * 64 + ci) * 7 + dk];
}

// ---------------------------------------------------------------------------
// WMMA implicit-GEMM conv(64->128, K=7) decomposed as 7 taps x K=64.
// Layer-1 activations (conv 2->64 + BN + lrelu) are recomputed on the fly:
// each thread owns one conv1 channel (weights in registers, tile-invariant)
// and 6 consecutive time steps (sliding x window), writing ONCE into a
// transposed, padded, double-buffered panel hT[tt][ci].  Per tile, ALL 14
// B fragments are loaded before an asm scheduling barrier (28 back-to-back
// ds_load_b128), then the 14 WMMAs issue as an uninterrupted chain.
// Per block: one batch, 1024 time steps = 64 tiles of 16.  8 waves = 8 M-tiles.
// ---------------------------------------------------------------------------
template <bool STORE_Y>
__global__ __launch_bounds__(256) void
k_conv2_wmma(const float* __restrict__ in,
             const float* __restrict__ W1k,
             const float* __restrict__ b1k,
             const float* __restrict__ s1k,
             const float* __restrict__ sh1k,
             const _Float16* __restrict__ afrag,
             const float* __restrict__ bias2,
             float* __restrict__ sumOut,
             float* __restrict__ sqOut,
             _Float16* __restrict__ ystore)
{
  __shared__ float    xs[2][1040];
  __shared__ _Float16 hT[2][22 * HSTR + 16];  // double-buffered activation panel

  const int tid  = threadIdx.x;
  const int wave = tid >> 5;
  const int lane = tid & 31;
  const int b    = blockIdx.x >> 7;           // 128 chunks of 1024 per batch
  const int base = (blockIdx.x & 127) * 1024;

  for (int v = tid; v < 2 * 1036; v += 256) {
    int i2 = v / 1036, j = v - i2 * 1036;
    int gt = base - 6 + j;
    xs[i2][j] = (gt >= 0 && gt < TLEN) ? in[((size_t)b * 2 + i2) * TLEN + gt] : 0.0f;
  }

  // build-phase ownership: channel bci, time steps btt0..btt0+5
  const int bci  = tid >> 2;
  const int btt0 = (tid & 3) * 6;
  float wreg[14];
#pragma unroll
  for (int q = 0; q < 14; ++q) wreg[q] = W1k[bci * 14 + q];
  const float bb   = b1k[bci];
  const float ssc  = s1k[bci];
  const float ssh  = sh1k[bci];

  // per-wave A fragments (weights), resident in VGPRs across all 64 tiles
  const v16h* ag = (const v16h*)afrag;
  v16h aF[14];
#pragma unroll
  for (int f = 0; f < 14; ++f) aF[f] = ag[(wave * 14 + f) * 32 + lane];

  const int g  = lane >> 4;
  const int n  = lane & 15;
  const int kb = g * 8;
  float b2loc[8];
#pragma unroll
  for (int r = 0; r < 8; ++r) b2loc[r] = bias2[wave * 16 + r + 8 * g];

  float accS[8], accQ[8];
#pragma unroll
  for (int r = 0; r < 8; ++r) { accS[r] = 0.0f; accQ[r] = 0.0f; }

  // branchless panel build: hT[buf][tt][bci] = h1[bci][t0 + tt - 3]
  auto build = [&](int tile, int buf) {
    const int xbase = tile * 16 + btt0;
    float x0[12], x1[12];
#pragma unroll
    for (int u = 0; u < 12; ++u) { x0[u] = xs[0][xbase + u]; x1[u] = xs[1][xbase + u]; }
#pragma unroll
    for (int u = 0; u < 6; ++u) {
      int tt = btt0 + u;
      float y = bb;
#pragma unroll
      for (int k = 0; k < 7; ++k)
        y += wreg[k] * x0[u + k] + wreg[7 + k] * x1[u + k];
      float h = ssc * y + ssh;
      h = (h >= 0.0f) ? h : 0.2f * h;
      int gt = base + tile * 16 + tt - 3;
      h = (gt >= 0 && gt < TLEN) ? h : 0.0f;   // SAME zero-padding of h1
      if (tt < 22) hT[buf][tt * HSTR + bci] = (_Float16)h;
    }
  };

  build(0, 0);
  __syncthreads();

  for (int tile = 0; tile < 64; ++tile) {
    const _Float16* rowp = &hT[tile & 1][n * HSTR + kb];

    // preload ALL 14 B fragments: 28 back-to-back ds_load_b128
    v16h bF[14];
#pragma unroll
    for (int f = 0; f < 14; ++f) {
      const _Float16* rp = rowp + (f >> 1) * HSTR + (f & 1) * 32;
      *reinterpret_cast<uint4*>(&bF[f])       = *reinterpret_cast<const uint4*>(rp);
      *(reinterpret_cast<uint4*>(&bF[f]) + 1) = *reinterpret_cast<const uint4*>(rp + 16);
    }
    // scheduling fence: LDS loads may not sink past this point, so the whole
    // load burst issues before the WMMA chain starts
    asm volatile("" ::: "memory");

    // uninterrupted WMMA accumulate chain
    v8f cacc = {0.f, 0.f, 0.f, 0.f, 0.f, 0.f, 0.f, 0.f};
#pragma unroll
    for (int f = 0; f < 14; ++f)
      cacc = __builtin_amdgcn_wmma_f32_16x16x32_f16(false, aF[f], false, bF[f],
                                                    (short)0, cacc, false, false);

    union { _Float16 h[8]; uint4 u4; } pk;
#pragma unroll
    for (int r = 0; r < 8; ++r) {
      float y = cacc[r] + b2loc[r];
      accS[r] += y;
      accQ[r] += y * y;
      pk.h[r] = (_Float16)y;
    }
    if (STORE_Y) {
      size_t gt = (size_t)base + tile * 16 + n;
      _Float16* p = ystore + ((size_t)b * TLEN + gt) * 128 + (wave * 16 + 8 * g);
      *reinterpret_cast<uint4*>(p) = pk.u4;
    }

    if (tile + 1 < 64) build(tile + 1, (tile + 1) & 1);  // overlaps with reads
    __syncthreads();
  }

  // reduce over time-columns (16 lanes per half-group), then global atomics
#pragma unroll
  for (int r = 0; r < 8; ++r) {
    for (int m = 1; m < 16; m <<= 1) {
      accS[r] += __shfl_xor(accS[r], m, 32);
      accQ[r] += __shfl_xor(accQ[r], m, 32);
    }
  }
  if ((lane & 15) == 0) {
#pragma unroll
    for (int r = 0; r < 8; ++r) {
      int ch = wave * 16 + r + 8 * g;
      atomicAdd(&sumOut[ch], accS[r]);
      atomicAdd(&sqOut[ch], accQ[r]);
    }
  }
}

// ---------------------------------------------------------------------------
// tiny head: recompute spatial-net layers at t=0..3, conv3 @ t=0, tanh,
// gather HRTF, produce F[b][i][c][k] = dist * hrtf  (one block per (b,i))
// ---------------------------------------------------------------------------
__global__ __launch_bounds__(128) void
k_head(const float* __restrict__ x,
       const float* W1, const float* b1,
       const float* s1, const float* sh1,
       const float* W2, const float* b2,
       const float* s2, const float* sh2,
       const float* W3, const float* b3,
       const float* hrtf, float* __restrict__ F)
{
  __shared__ float h1s[64][10];
  __shared__ float h2n[128][4];
  __shared__ float spv[3];
  __shared__ int   sidx;
  __shared__ float sdist;
  const int tid = threadIdx.x;
  const int b = blockIdx.x >> 1, i = blockIdx.x & 1;

  if (tid < 64) {
    int ci = tid;
    for (int tt = 0; tt < 10; ++tt) {
      int t = tt - 3;
      float h = 0.0f;
      if (t >= 0) {
        float y = b1[ci];
        for (int i2 = 0; i2 < 2; ++i2)
          for (int k = 0; k < 7; ++k) {
            int gt = t + k - 3;
            if (gt >= 0 && gt < TLEN)
              y += W1[(ci * 2 + i2) * 7 + k] * x[((size_t)b * 2 + i2) * TLEN + gt];
          }
        h = s1[ci] * y + sh1[ci];
        h = (h >= 0.0f) ? h : 0.2f * h;
      }
      h1s[ci][tt] = h;
    }
  }
  __syncthreads();
  {
    int co = tid;   // 128 threads
    for (int t = 0; t < 4; ++t) {
      float y = b2[co];
      for (int ci = 0; ci < 64; ++ci)
        for (int dk = 0; dk < 7; ++dk)
          y += W2[(co * 64 + ci) * 7 + dk] * h1s[ci][t + dk];
      float h = s2[co] * y + sh2[co];
      h2n[co][t] = (h >= 0.0f) ? h : 0.2f * h;
    }
  }
  __syncthreads();
  if (tid < 3) {
    int c3 = i * 3 + tid;
    float acc = b3[c3];
    for (int dk = 3; dk < 7; ++dk)
      for (int ci = 0; ci < 128; ++ci)
        acc += W3[(c3 * 128 + ci) * 7 + dk] * h2n[ci][dk - 3];
    spv[tid] = tanhf(acc);
  }
  __syncthreads();
  if (tid == 0) {
    float az = (spv[0] + 1.0f) * 0.5f * 11.0f;
    float el = (spv[1] + 1.0f) * 0.5f * 4.0f;
    float d  = (spv[2] + 1.0f) * 0.45f + 0.1f;
    int ai = (int)az; ai = ai < 0 ? 0 : (ai > 11 ? 11 : ai);
    int ei = (int)el; ei = ei < 0 ? 0 : (ei > 4 ? 4 : ei);
    sidx = ei * 12 + ai;
    sdist = d;
  }
  __syncthreads();
  if (tid < 14) {
    int c = tid / 7, k = tid - c * 7;
    F[(((size_t)b * 2 + i) * 2 + c) * 7 + k] = sdist * hrtf[(sidx * 2 + c) * 7 + k];
  }
}

// ---------------------------------------------------------------------------
// summed[b][c][t] = sum_i conv(x[b][i], F[b][i][c]) — memory bound
// ---------------------------------------------------------------------------
__global__ __launch_bounds__(256) void
k_mix(const float* __restrict__ x, const float* __restrict__ F,
      float* __restrict__ summed)
{
  int gid = blockIdx.x * 256 + threadIdx.x;
  int b = gid / TLEN;
  int t = gid % TLEN;
  float a0 = 0.0f, a1 = 0.0f;
#pragma unroll
  for (int i2 = 0; i2 < 2; ++i2) {
#pragma unroll
    for (int k = 0; k < 7; ++k) {
      int tt = t + k - 3;
      float xv = (tt >= 0 && tt < TLEN) ? x[((size_t)b * 2 + i2) * TLEN + tt] : 0.0f;
      a0 += F[(((size_t)b * 2 + i2) * 2 + 0) * 7 + k] * xv;
      a1 += F[(((size_t)b * 2 + i2) * 2 + 1) * 7 + k] * xv;
    }
  }
  summed[((size_t)b * 2 + 0) * TLEN + t] = a0;
  summed[((size_t)b * 2 + 1) * TLEN + t] = a1;
}

// ---------------------------------------------------------------------------
// final layer: BN + lrelu on y2r (f16), conv(128->2, K=7), tanh -> out
// ---------------------------------------------------------------------------
__global__ __launch_bounds__(256) void
k_r3(const _Float16* __restrict__ y2r,
     const float* __restrict__ scl, const float* __restrict__ shf,
     const float* __restrict__ RW3, const float* __restrict__ rb3,
     float* __restrict__ out)
{
  __shared__ _Float16 rows[262 * 128];
  __shared__ float    wr3s[2 * 128 * 7];
  __shared__ float    sc[128], sh[128];
  const int tid  = threadIdx.x;
  const int b    = blockIdx.x >> 9;           // 512 chunks of 256 per batch
  const int base = (blockIdx.x & 511) * 256;

  for (int v = tid; v < 2 * 128 * 7; v += 256) wr3s[v] = RW3[v];
  if (tid < 128) { sc[tid] = scl[tid]; sh[tid] = shf[tid]; }
  __syncthreads();
  for (int v = tid; v < 262 * 128; v += 256) {
    int row = v >> 7, ci = v & 127;
    int gt = base - 3 + row;
    float f = 0.0f;
    if (gt >= 0 && gt < TLEN) {
      f = (float)y2r[((size_t)b * TLEN + gt) * 128 + ci];
      f = sc[ci] * f + sh[ci];
      f = (f >= 0.0f) ? f : 0.2f * f;
    }
    rows[v] = (_Float16)f;
  }
  __syncthreads();

  float a0 = 0.0f, a1 = 0.0f;
#pragma unroll
  for (int dk = 0; dk < 7; ++dk) {
    int rb = (tid + dk) << 7;
    for (int ci = 0; ci < 128; ci += 2) {
      v2h pv = *reinterpret_cast<const v2h*>(&rows[rb + ci]);
      float v0 = (float)pv.x, v1 = (float)pv.y;
      a0 += wr3s[ci * 7 + dk] * v0 + wr3s[(ci + 1) * 7 + dk] * v1;
      a1 += wr3s[896 + ci * 7 + dk] * v0 + wr3s[896 + (ci + 1) * 7 + dk] * v1;
    }
  }
  int t = base + tid;
  out[((size_t)b * 2 + 0) * TLEN + t] = tanhf(a0 + rb3[0]);
  out[((size_t)b * 2 + 1) * TLEN + t] = tanhf(a1 + rb3[1]);
}

// ---------------------------------------------------------------------------
extern "C" void kernel_launch(void* const* d_in, const int* in_sizes, int n_in,
                              void* d_out, int out_size, void* d_ws, size_t ws_size,
                              hipStream_t stream)
{
  const float* x    = (const float*)d_in[0];
  const float* W1   = (const float*)d_in[1];
  const float* b1   = (const float*)d_in[2];
  const float* g1   = (const float*)d_in[3];
  const float* be1  = (const float*)d_in[4];
  const float* W2   = (const float*)d_in[5];
  const float* b2   = (const float*)d_in[6];
  const float* g2   = (const float*)d_in[7];
  const float* be2  = (const float*)d_in[8];
  const float* W3   = (const float*)d_in[9];
  const float* b3   = (const float*)d_in[10];
  const float* hrtf = (const float*)d_in[11];
  const float* RW1  = (const float*)d_in[12];
  const float* rb1  = (const float*)d_in[13];
  const float* rg1  = (const float*)d_in[14];
  const float* rbe1 = (const float*)d_in[15];
  const float* RW2  = (const float*)d_in[16];
  const float* rb2  = (const float*)d_in[17];
  const float* rg2  = (const float*)d_in[18];
  const float* rbe2 = (const float*)d_in[19];
  const float* RW3  = (const float*)d_in[20];
  const float* rb3  = (const float*)d_in[21];
  float* out = (float*)d_out;

  char* ws = (char*)d_ws;
  size_t off = 0;
  auto take = [&](size_t bytes) -> char* {
    char* p = ws + off;
    off = (off + bytes + 255) & ~(size_t)255;
    return p;
  };
  float* stats = (float*)take(768 * sizeof(float));   // zeroed each launch
  float* sum1 = stats,        *sq1 = stats + 64;
  float* sum2 = stats + 128,  *sq2 = stats + 256;
  float* sumR1 = stats + 384, *sqR1 = stats + 448;
  float* sumR2 = stats + 512, *sqR2 = stats + 640;
  float* scale1  = (float*)take(64 * 4);  float* shift1  = (float*)take(64 * 4);
  float* scale2  = (float*)take(128 * 4); float* shift2  = (float*)take(128 * 4);
  float* scaleR1 = (float*)take(64 * 4);  float* shiftR1 = (float*)take(64 * 4);
  float* scaleR2 = (float*)take(128 * 4); float* shiftR2 = (float*)take(128 * 4);
  float* Fw = (float*)take(224 * 4);
  _Float16* A2  = (_Float16*)take(57344 * 2);
  _Float16* AR2 = (_Float16*)take(57344 * 2);
  float* summed = (float*)take((size_t)NB * 2 * TLEN * 4);
  _Float16* y2r = (_Float16*)take((size_t)NB * TLEN * 128 * 2);
  if (off > ws_size) return;

  const float invN = 1.0f / (8.0f * 131072.0f);

  hipMemsetAsync(stats, 0, 768 * sizeof(float), stream);
  k_pack_wfrag<<<224, 256, 0, stream>>>(W2, A2);
  k_pack_wfrag<<<224, 256, 0, stream>>>(RW2, AR2);

  // ---- spatial param net ----
  k_conv_stats<<<512, 256, 0, stream>>>(x, W1, b1, sum1, sq1);
  k_finalize<<<1, 128, 0, stream>>>(sum1, sq1, g1, be1, scale1, shift1, 64, invN);
  k_conv2_wmma<false><<<1024, 256, 0, stream>>>(x, W1, b1, scale1, shift1,
                                                A2, b2, sum2, sq2, nullptr);
  k_finalize<<<1, 128, 0, stream>>>(sum2, sq2, g2, be2, scale2, shift2, 128, invN);
  k_head<<<16, 128, 0, stream>>>(x, W1, b1, scale1, shift1,
                                 W2, b2, scale2, shift2, W3, b3, hrtf, Fw);

  // ---- HRTF mix ----
  k_mix<<<4096, 256, 0, stream>>>(x, Fw, summed);

  // ---- binaural renderer ----
  k_conv_stats<<<512, 256, 0, stream>>>(summed, RW1, rb1, sumR1, sqR1);
  k_finalize<<<1, 128, 0, stream>>>(sumR1, sqR1, rg1, rbe1, scaleR1, shiftR1, 64, invN);
  k_conv2_wmma<true><<<1024, 256, 0, stream>>>(summed, RW1, rb1, scaleR1, shiftR1,
                                               AR2, rb2, sumR2, sqR2, y2r);
  k_finalize<<<1, 128, 0, stream>>>(sumR2, sqR2, rg2, rbe2, scaleR2, shiftR2, 128, invN);
  k_r3<<<4096, 256, 0, stream>>>(y2r, scaleR2, shiftR2, RW3, rb3, out);
}
